// HopAggLayer_39676907888226
// MI455X (gfx1250) — matrix operand order repaired and checked by reading, not compiled, not run on hardware
//
#include <hip/hip_runtime.h>

// ---------------------------------------------------------------------------
// CDNA5 / gfx1250 fused TGN-style hop-aggregation kernel.
// wave32, WMMA bf16 16x16x32, f32 accumulate, double-buffered LDS staging.
// ---------------------------------------------------------------------------

typedef __attribute__((ext_vector_type(16))) __bf16 v16bf;
typedef __attribute__((ext_vector_type(8)))  __bf16 bf16x8;
typedef __attribute__((ext_vector_type(4)))  __bf16 bf16x4;
typedef __attribute__((ext_vector_type(8)))  float  v8f;

#define DDIM 128
#define KN   40      // k * n_neighbors
#define AS   400     // bf16 row stride of staged A tile (800B = 200 dwords, %64=8)
#define SS   136     // f32 row stride of S tiles (%64 = 8)

__device__ inline v8f zero8() {
    v8f z;
#pragma unroll
    for (int i = 0; i < 8; ++i) z[i] = 0.0f;
    return z;
}

// 16x32 bf16 A-fragment from an LDS row (lane<16: K=kb..kb+7, kb+16..kb+23;
// lanes 16-31 shifted by +8).  Two 16B DS loads.
__device__ inline v16bf load_afrag_bf16(const __bf16* Arow, int kb) {
    bf16x8 lo = *(const bf16x8*)(Arow + kb);
    bf16x8 hi = *(const bf16x8*)(Arow + kb + 16);
    v16bf a;
#pragma unroll
    for (int i = 0; i < 8; ++i) { a[i] = lo[i]; a[8 + i] = hi[i]; }
    return a;
}

// 16x32 A-fragment from an f32 LDS row (convert on the fly).
__device__ inline v16bf load_afrag_f32(const float* Arow, int kb) {
    v16bf a;
#pragma unroll
    for (int i = 0; i < 8; ++i) {
        a[i]     = (__bf16)Arow[kb + i];
        a[8 + i] = (__bf16)Arow[kb + 16 + i];
    }
    return a;
}

// 32x16 bf16 B-fragment from a row-major f32 weight [*,128].
// lane<16: col n, K rows k0+v, k0+16+v; lanes 16-31: K rows +8.
__device__ inline v16bf load_bfrag(const float* __restrict__ W, int k0, int n) {
    v16bf b;
#pragma unroll
    for (int v = 0; v < 8; ++v) {
        b[v]     = (__bf16)W[(k0 + v) * DDIM + n];
        b[8 + v] = (__bf16)W[(k0 + 16 + v) * DDIM + n];
    }
    return b;
}

// Cooperatively stage cat = [nbr | time | edge] (16 rows x 384 f32) for
// neighbor n into LDS as bf16.  256 threads, float4 granularity.
__device__ inline void load_cat_tile(__bf16* dst,
                                     const float* __restrict__ nf,
                                     const float* __restrict__ tf,
                                     const float* __restrict__ ef,
                                     long b0, int n, int tid) {
#pragma unroll
    for (int i = 0; i < 6; ++i) {
        int idx    = tid + i * 256;     // 0..1535 float4 slots
        int r      = idx / 96;          // batch row within tile
        int w4     = idx % 96;
        int region = w4 >> 5;           // 0=nbr, 1=time, 2=edge
        int c4     = (w4 & 31) * 4;
        const float* src = (region == 0) ? nf : ((region == 1) ? tf : ef);
        const float4 v = *(const float4*)(src + ((b0 + r) * KN + n) * DDIM + c4);
        bf16x4 p;
        p[0] = (__bf16)v.x; p[1] = (__bf16)v.y;
        p[2] = (__bf16)v.z; p[3] = (__bf16)v.w;
        *(bf16x4*)(dst + r * AS + region * DDIM + c4) = p;
    }
}

__global__ __launch_bounds__(256, 1)
void hopagg_kernel(const float* __restrict__ srcf,   // [B,128]
                   const float* __restrict__ srct,   // [B,1,128] -> [B,128]
                   const float* __restrict__ nf,     // [B,40,128]
                   const float* __restrict__ tf,     // [B,40,128]
                   const float* __restrict__ ef,     // [B,40,128]
                   const float* __restrict__ Wnbr,   // [384,128]
                   const float* __restrict__ bnbr,   // [128]
                   const float* __restrict__ Wsrc,   // [256,128]
                   const float* __restrict__ bsrc,   // [128]
                   const float* __restrict__ Wcat,   // [512,128] = 4 x [128,128]
                   const float* __restrict__ bcat,   // [128]
                   const int*   __restrict__ maskp,  // [B,40] nonzero => zero row
                   float* __restrict__ out)          // [B,128]
{
    __shared__ __bf16 A_lds[2][16 * AS];   // double-buffered cat tile (bf16)
    __shared__ float  S_lds[4][16 * SS];   // residue-class sums (f32)
    __shared__ float  mmul[16 * KN];       // 0/1 mask multipliers

    const int  tid  = threadIdx.x;
    const int  lane = tid & 31;
    const int  wv   = tid >> 5;            // wave id == output col-tile 0..7
    const long b0   = (long)blockIdx.x * 16;

    const int mrow = lane & 15;            // A row / output col within tile
    const int half = lane >> 4;            // wave half select
    const int cl   = wv * 16 + mrow;       // output column 0..127

    const float bnbr_c = bnbr[cl];
    const float bsrc_c = bsrc[cl];
    const float bcat_c = bcat[cl];

    // ---- stage mask multipliers -------------------------------------------
    for (int i = tid; i < 16 * KN; i += 256) {
        int r = i / KN, nn = i % KN;
        mmul[i] = maskp[(b0 + r) * KN + nn] ? 0.0f : 1.0f;
    }

    // ---- preload W_nbr B-fragments for this wave's col-tile (held in VGPRs)
    v16bf bw[12];
#pragma unroll
    for (int ks = 0; ks < 12; ++ks)
        bw[ks] = load_bfrag(Wnbr, ks * 32 + (half << 3), cl);

    v8f accS[4];
#pragma unroll
    for (int j = 0; j < 4; ++j) accS[j] = zero8();

    // ---- neighbor loop: GEMM [16,384]x[384,128] per neighbor, masked sum --
    // Unrolled by 4 so the residue-class accumulator index is compile-time
    // constant (avoids indexed-VGPR movrel access).
    load_cat_tile(A_lds[0], nf, tf, ef, b0, 0, tid);
    __syncthreads();

    for (int n4 = 0; n4 < KN; n4 += 4) {
#pragma unroll
        for (int jj = 0; jj < 4; ++jj) {
            const int n = n4 + jj;
            if (n + 1 < KN)
                load_cat_tile(A_lds[(n + 1) & 1], nf, tf, ef, b0, n + 1, tid);

            const __bf16* Arow = A_lds[n & 1] + mrow * AS;
            v8f y = zero8();
#pragma unroll
            for (int ks = 0; ks < 12; ++ks) {
                v16bf a = load_afrag_bf16(Arow, ks * 32 + (half << 3));
                y = __builtin_amdgcn_wmma_f32_16x16x32_bf16(
                        false, a, false, bw[ks], (short)0, y, false, false);
            }

#pragma unroll
            for (int e = 0; e < 8; ++e) {
                int   m     = e + (half << 3);
                float mult  = mmul[m * KN + n];
                // residual = nbr feature, re-read from the staged bf16 tile
                // (region 0) -> no second HBM pass over neighbors_features.
                float resid = (float)A_lds[n & 1][m * AS + cl];
                accS[jj][e] += mult * (y[e] + bnbr_c + resid);
            }
            __syncthreads();
        }
    }

    // ---- spill residue-class sums S_j[16,128] to LDS ----------------------
#pragma unroll
    for (int j = 0; j < 4; ++j)
#pragma unroll
        for (int e = 0; e < 8; ++e) {
            int m = e + (half << 3);
            S_lds[j][m * SS + cl] = accS[j][e];
        }

    // ---- stage src tile [16 x 256] (reuse A buffer 0) ---------------------
#pragma unroll
    for (int i = 0; i < 4; ++i) {
        int idx    = tid + i * 256;   // 0..1023 float4 slots
        int r      = idx >> 6;
        int w4     = idx & 63;
        int region = w4 >> 5;         // 0 = src features, 1 = src time
        int c4     = (w4 & 31) * 4;
        const float* src = region ? srct : srcf;
        const float4 v = *(const float4*)(src + (b0 + r) * DDIM + c4);
        bf16x4 p;
        p[0] = (__bf16)v.x; p[1] = (__bf16)v.y;
        p[2] = (__bf16)v.z; p[3] = (__bf16)v.w;
        *(bf16x4*)(A_lds[0] + r * AS + region * DDIM + c4) = p;
    }
    __syncthreads();

    // ---- epilogue GEMMs: new_src + sum_j S_j @ Wc[j] ----------------------
    v8f acc = zero8();

    // new_src: [16,256] @ W_src[256,128]
    {
        const __bf16* Arow = A_lds[0] + mrow * AS;
#pragma unroll
        for (int ks = 0; ks < 8; ++ks) {
            v16bf a = load_afrag_bf16(Arow, ks * 32 + (half << 3));
            v16bf b = load_bfrag(Wsrc, ks * 32 + (half << 3), cl);
            acc = __builtin_amdgcn_wmma_f32_16x16x32_bf16(
                      false, a, false, b, (short)0, acc, false, false);
        }
    }

    // neighbor_tensor: sum_j S_j[16,128] @ W_cat3[j*128 : (j+1)*128, :]
#pragma unroll
    for (int j = 0; j < 4; ++j) {
        const float* Arow = &S_lds[j][mrow * SS];
#pragma unroll
        for (int ks = 0; ks < 4; ++ks) {
            v16bf a = load_afrag_f32(Arow, ks * 32 + (half << 3));
            v16bf b = load_bfrag(Wcat, j * 128 + ks * 32 + (half << 3), cl);
            acc = __builtin_amdgcn_wmma_f32_16x16x32_bf16(
                      false, a, false, b, (short)0, acc, false, false);
        }
    }

    // out = new_src + neighbor_tensor + b_src + 10*b_cat3
#pragma unroll
    for (int e = 0; e < 8; ++e) {
        int m = e + (half << 3);
        out[(b0 + m) * DDIM + cl] = acc[e] + bsrc_c + 10.0f * bcat_c;
    }
}

// ---------------------------------------------------------------------------
extern "C" void kernel_launch(void* const* d_in, const int* in_sizes, int n_in,
                              void* d_out, int out_size, void* d_ws, size_t ws_size,
                              hipStream_t stream) {
    const float* srcf = (const float*)d_in[0];
    const float* srct = (const float*)d_in[1];
    const float* nf   = (const float*)d_in[2];
    const float* tf   = (const float*)d_in[3];
    const float* ef   = (const float*)d_in[4];
    const float* Wnbr = (const float*)d_in[5];
    const float* bnbr = (const float*)d_in[6];
    const float* Wsrc = (const float*)d_in[7];
    const float* bsrc = (const float*)d_in[8];
    const float* Wcat = (const float*)d_in[9];
    const float* bcat = (const float*)d_in[10];
    const int*   mask = (const int*)d_in[11];

    const int B = in_sizes[0] / DDIM;   // 16384
    dim3 grid(B / 16), block(256);
    hipLaunchKernelGGL(hopagg_kernel, grid, block, 0, stream,
                       srcf, srct, nf, tf, ef, Wnbr, bnbr, Wsrc, bsrc,
                       Wcat, bcat, mask, (float*)d_out);
}
